// Propagate_78898549227884
// MI455X (gfx1250) — matrix operand order repaired and checked by reading, not compiled, not run on hardware
//
#include <hip/hip_runtime.h>
#include <hip/hip_bf16.h>
#include <math.h>

typedef __attribute__((ext_vector_type(16))) __bf16 v16bf;
typedef __attribute__((ext_vector_type(8)))  float  v8f;

#define Bc 4
#define Nc 16384
#define Mc 2048
#define BN (Bc * Nc)          // 65536 points
#define LN_EPS 1e-5f

union Frag16 { uint4 u[2]; v16bf v; };

__device__ __forceinline__ unsigned short bf2u(__bf16 x) {
  union { __bf16 b; unsigned short u; } c; c.b = x; return c.u;
}
__device__ __forceinline__ unsigned int pk2(float a, float b) {
  return (unsigned int)bf2u((__bf16)a) | ((unsigned int)bf2u((__bf16)b) << 16);
}

// ---------------------------------------------------------------------------
// weights: f32 row-major [k][n] -> bf16 TRANSPOSED [n][k]  (one-shot, 64KB)
// Transposed layout makes every WMMA B-fragment two 16B contiguous global runs.
// ---------------------------------------------------------------------------
__global__ __launch_bounds__(256) void cvt_bf16_t_kernel(const float* __restrict__ src,
                                                         __bf16* __restrict__ dst) {
  const int i = blockIdx.x * 256 + threadIdx.x;   // 65536 elements
  const int k = i >> 8, n = i & 255;
  dst[n * 256 + k] = (__bf16)src[i];              // src[i] == src[k*256+n]
}

// ---------------------------------------------------------------------------
// KNN(3): thread = query point; centers streamed through LDS (full reuse)
// grid = (N/256, B), block = 256
// ---------------------------------------------------------------------------
__global__ __launch_bounds__(256) void knn3_kernel(const float* __restrict__ xyz,
                                                   const float* __restrict__ centers,
                                                   int* __restrict__ idxOut,
                                                   float* __restrict__ wOut) {
  __shared__ float sc[256 * 3];
  const int b = blockIdx.y;
  const int n = blockIdx.x * 256 + threadIdx.x;
  const size_t p = (size_t)b * Nc + n;
  const float px = xyz[p * 3 + 0];
  const float py = xyz[p * 3 + 1];
  const float pz = xyz[p * 3 + 2];

  float d0 = 3.0e38f, d1 = 3.0e38f, d2 = 3.0e38f;
  int   i0 = 0, i1 = 0, i2 = 0;

  for (int c0 = 0; c0 < Mc; c0 += 256) {
    __syncthreads();
    const float* csrc = centers + ((size_t)b * Mc + c0) * 3;
    for (int t = threadIdx.x; t < 256 * 3; t += 256) sc[t] = csrc[t];
    __syncthreads();
    #pragma unroll 4
    for (int j = 0; j < 256; ++j) {
      const float dx = sc[j * 3 + 0] - px;
      const float dy = sc[j * 3 + 1] - py;
      const float dz = sc[j * 3 + 2] - pz;
      const float d = dx * dx + dy * dy + dz * dz;
      const int ci = c0 + j;
      if (d < d2) {
        if (d < d0)      { d2 = d1; i2 = i1; d1 = d0; i1 = i0; d0 = d; i0 = ci; }
        else if (d < d1) { d2 = d1; i2 = i1; d1 = d;  i1 = ci; }
        else             { d2 = d;  i2 = ci; }
      }
    }
  }
  const float r0 = 1.0f / (fmaxf(d0, 0.0f) + 1e-8f);
  const float r1 = 1.0f / (fmaxf(d1, 0.0f) + 1e-8f);
  const float r2 = 1.0f / (fmaxf(d2, 0.0f) + 1e-8f);
  const float rs = 1.0f / (r0 + r1 + r2);
  idxOut[p * 3 + 0] = i0;  wOut[p * 3 + 0] = r0 * rs;
  idxOut[p * 3 + 1] = i1;  wOut[p * 3 + 1] = r1 * rs;
  idxOut[p * 3 + 2] = i2;  wOut[p * 3 + 2] = r2 * rs;
}

// ---------------------------------------------------------------------------
// rel layer-1 (3->256) + LN + GELU, weighted-combined over K.
// Wave per point; lane owns 8 CONTIGUOUS cols (float4 loads, uint4 bf16 store)
// ---------------------------------------------------------------------------
__global__ __launch_bounds__(256) void rel_hidden_kernel(
    const float* __restrict__ xyz, const float* __restrict__ centers,
    const int* __restrict__ idx, const float* __restrict__ wgt,
    const float* __restrict__ W1, const float* __restrict__ b1,
    const float* __restrict__ g, const float* __restrict__ bb,
    __bf16* __restrict__ G) {
  const int lane = threadIdx.x & 31;
  const int c0 = lane * 8;
  int wv = (blockIdx.x * 256 + threadIdx.x) >> 5;
  const int nW = gridDim.x * 8;

  float w1x[8], w1y[8], w1z[8], bv[8], gv[8], bbv[8];
  #pragma unroll
  for (int h = 0; h < 2; ++h) {
    *(float4*)(w1x + 4*h) = *(const float4*)(W1 +       c0 + 4*h);
    *(float4*)(w1y + 4*h) = *(const float4*)(W1 + 256 + c0 + 4*h);
    *(float4*)(w1z + 4*h) = *(const float4*)(W1 + 512 + c0 + 4*h);
    *(float4*)(bv  + 4*h) = *(const float4*)(b1 + c0 + 4*h);
    *(float4*)(gv  + 4*h) = *(const float4*)(g  + c0 + 4*h);
    *(float4*)(bbv + 4*h) = *(const float4*)(bb + c0 + 4*h);
  }

  for (int p = wv; p < BN; p += nW) {
    const int b = p >> 14;   // N = 16384
    const float px = xyz[(size_t)p * 3 + 0];
    const float py = xyz[(size_t)p * 3 + 1];
    const float pz = xyz[(size_t)p * 3 + 2];
    float acc[8];
    #pragma unroll
    for (int j = 0; j < 8; ++j) acc[j] = 0.0f;

    for (int k = 0; k < 3; ++k) {
      const int   ci = idx[(size_t)p * 3 + k];
      const float wk = wgt[(size_t)p * 3 + k];
      const float* cc = centers + ((size_t)b * Mc + ci) * 3;
      const float rx = cc[0] - px, ry = cc[1] - py, rz = cc[2] - pz;
      float h[8], s = 0.0f, s2 = 0.0f;
      #pragma unroll
      for (int j = 0; j < 8; ++j) {
        const float v = rx * w1x[j] + ry * w1y[j] + rz * w1z[j] + bv[j];
        h[j] = v; s += v; s2 += v * v;
      }
      #pragma unroll
      for (int off = 16; off > 0; off >>= 1) {
        s  += __shfl_xor(s, off, 32);
        s2 += __shfl_xor(s2, off, 32);
      }
      const float mu  = s * (1.0f / 256.0f);
      const float var = s2 * (1.0f / 256.0f) - mu * mu;
      const float inv = rsqrtf(var + LN_EPS);
      #pragma unroll
      for (int j = 0; j < 8; ++j) {
        const float xn = (h[j] - mu) * inv * gv[j] + bbv[j];
        acc[j] += wk * (0.5f * xn * (1.0f + erff(xn * 0.70710678f)));
      }
    }
    uint4 outp;
    outp.x = pk2(acc[0], acc[1]); outp.y = pk2(acc[2], acc[3]);
    outp.z = pk2(acc[4], acc[5]); outp.w = pk2(acc[6], acc[7]);
    *(uint4*)(G + (size_t)p * 256 + c0) = outp;
  }
}

// ---------------------------------------------------------------------------
// bf16 WMMA GEMM: C[rows x 256] = A[rows x 256] @ W[256 x 256] + bias
// Wt is PRE-TRANSPOSED bf16 [n][k]: each lane's B fragment = two 16B
// global runs 32B apart -> 2x global_load_b128. A tile (32x256) in LDS; A fragment =
// 2x ds_load_b128. 8 waves: rowGroup = (w>>2)*16, colBase = (w&3)*64.
// Fully unrolled K loop: 32 v_wmma_f32_16x16x32_bf16 per wave.
// ---------------------------------------------------------------------------
__global__ __launch_bounds__(256) void gemm_bf16_256(
    const __bf16* __restrict__ A, const __bf16* __restrict__ Wt,
    const float* __restrict__ bias, float* __restrict__ C, int rows) {
  __shared__ __align__(16) __bf16 sA[32 * 256];
  const int tid  = threadIdx.x;
  const int wave = tid >> 5;
  const int lane = tid & 31;
  const int rowBlk = blockIdx.x * 32;

  { // stage A tile: 32x256 bf16 = 1024 uint4
    const uint4* src = (const uint4*)(A + (size_t)rowBlk * 256);
    uint4* dst = (uint4*)sA;
    #pragma unroll
    for (int i = 0; i < 4; ++i) dst[tid + 256 * i] = src[tid + 256 * i];
  }
  __syncthreads();

  const int rowGroup = (wave >> 2) * 16;  // 0 | 16
  const int colBase  = (wave & 3) * 64;   // 0,64,128,192
  const int half = lane >> 4;
  const int ml   = lane & 15;

  // A: per-lane K pattern = {0..7,16..23} (half 0) / {8..15,24..31} (half 1)
  const __bf16* arow = sA + (rowGroup + ml) * 256 + half * 8;
  // B: lane reads Wt[col][kt*32 + half*8 + {0..7,16..23}] : 2x 16B runs
  const __bf16* wcol0 = Wt + (size_t)(colBase + ml) * 256 + half * 8;

  v8f acc[4];
  #pragma unroll
  for (int nt = 0; nt < 4; ++nt)
    acc[nt] = (v8f){0.f,0.f,0.f,0.f,0.f,0.f,0.f,0.f};

  #pragma unroll
  for (int kt = 0; kt < 8; ++kt) {
    Frag16 fa;
    fa.u[0] = *(const uint4*)(arow + kt * 32);
    fa.u[1] = *(const uint4*)(arow + kt * 32 + 16);
    #pragma unroll
    for (int nt = 0; nt < 4; ++nt) {
      Frag16 fb;
      const __bf16* bp = wcol0 + (size_t)nt * 16 * 256 + kt * 32;
      fb.u[0] = *(const uint4*)(bp);
      fb.u[1] = *(const uint4*)(bp + 16);
      acc[nt] = __builtin_amdgcn_wmma_f32_16x16x32_bf16(
          false, fa.v, false, fb.v, (short)0, acc[nt], false, false);
    }
  }

  // C layout: VGPR r -> lanes 0-15: M=r, lanes 16-31: M=8+r
  #pragma unroll
  for (int nt = 0; nt < 4; ++nt) {
    const int col = colBase + nt * 16 + ml;
    const float bvv = bias[col];
    #pragma unroll
    for (int r = 0; r < 8; ++r) {
      const int row = rowBlk + rowGroup + half * 8 + r;
      C[(size_t)row * 256 + col] = acc[nt][r] + bvv;
    }
  }
  (void)rows;
}

// ---------------------------------------------------------------------------
// skip = skipPre + sum_k w_k * center_feats[idx_k]; also emit bf16 copy
// grid = BN, block = 256 (one col per thread, coalesced gathers)
// ---------------------------------------------------------------------------
__global__ __launch_bounds__(256) void combine_kernel(
    const float* __restrict__ cf, const int* __restrict__ idx,
    const float* __restrict__ wgt, float* __restrict__ skip,
    __bf16* __restrict__ skipb) {
  const int p = blockIdx.x, c = threadIdx.x;
  const int b = p >> 14;
  float acc = skip[(size_t)p * 256 + c];
  #pragma unroll
  for (int k = 0; k < 3; ++k) {
    const int ci = idx[(size_t)p * 3 + k];
    acc += wgt[(size_t)p * 3 + k] * cf[((size_t)b * Mc + ci) * 256 + c];
  }
  skip [(size_t)p * 256 + c] = acc;
  skipb[(size_t)p * 256 + c] = (__bf16)acc;
}

// ---------------------------------------------------------------------------
// h += rgb @ mlp_W1[256:259, :]  (rank-3 tail of the concat GEMM)
// ---------------------------------------------------------------------------
__global__ __launch_bounds__(256) void rgb_tail_kernel(
    const float* __restrict__ rgb, const float* __restrict__ W1,
    float* __restrict__ h) {
  const int p = blockIdx.x, c = threadIdx.x;
  const float r0 = rgb[(size_t)p * 3 + 0];
  const float r1 = rgb[(size_t)p * 3 + 1];
  const float r2 = rgb[(size_t)p * 3 + 2];
  h[(size_t)p * 256 + c] += r0 * W1[256 * 256 + c]
                          + r1 * W1[257 * 256 + c]
                          + r2 * W1[258 * 256 + c];
}

// ---------------------------------------------------------------------------
// LN + exact GELU over 256-wide rows; f32 in, bf16 out. Wave per row,
// lane owns 8 contiguous cols: 2x float4 in, 1x uint4 (8xbf16) out.
// ---------------------------------------------------------------------------
__global__ __launch_bounds__(256) void ln_gelu_kernel(
    const float* __restrict__ X, const float* __restrict__ g,
    const float* __restrict__ bb, __bf16* __restrict__ Y) {
  const int lane = threadIdx.x & 31;
  const int c0 = lane * 8;
  int wv = (blockIdx.x * 256 + threadIdx.x) >> 5;
  const int nW = gridDim.x * 8;

  float gv[8], bbv[8];
  #pragma unroll
  for (int h = 0; h < 2; ++h) {
    *(float4*)(gv  + 4*h) = *(const float4*)(g  + c0 + 4*h);
    *(float4*)(bbv + 4*h) = *(const float4*)(bb + c0 + 4*h);
  }

  for (int p = wv; p < BN; p += nW) {
    float h[8];
    *(float4*)(h + 0) = *(const float4*)(X + (size_t)p * 256 + c0);
    *(float4*)(h + 4) = *(const float4*)(X + (size_t)p * 256 + c0 + 4);
    float s = 0.0f, s2 = 0.0f;
    #pragma unroll
    for (int j = 0; j < 8; ++j) { s += h[j]; s2 += h[j] * h[j]; }
    #pragma unroll
    for (int off = 16; off > 0; off >>= 1) {
      s  += __shfl_xor(s, off, 32);
      s2 += __shfl_xor(s2, off, 32);
    }
    const float mu  = s * (1.0f / 256.0f);
    const float var = s2 * (1.0f / 256.0f) - mu * mu;
    const float inv = rsqrtf(var + LN_EPS);
    float o[8];
    #pragma unroll
    for (int j = 0; j < 8; ++j) {
      const float xn = (h[j] - mu) * inv * gv[j] + bbv[j];
      o[j] = 0.5f * xn * (1.0f + erff(xn * 0.70710678f));
    }
    uint4 outp;
    outp.x = pk2(o[0], o[1]); outp.y = pk2(o[2], o[3]);
    outp.z = pk2(o[4], o[5]); outp.w = pk2(o[6], o[7]);
    *(uint4*)(Y + (size_t)p * 256 + c0) = outp;
  }
}

// ---------------------------------------------------------------------------
// y_bf16 = a + b (residual); 4 elems/thread, uint2 packed store
// ---------------------------------------------------------------------------
__global__ __launch_bounds__(256) void residual_kernel(
    const float* __restrict__ a, const float* __restrict__ b,
    __bf16* __restrict__ y) {
  const size_t i = ((size_t)blockIdx.x * 256 + threadIdx.x) * 4;
  const float4 av = *(const float4*)(a + i);
  const float4 bv = *(const float4*)(b + i);
  uint2 outp;
  outp.x = pk2(av.x + bv.x, av.y + bv.y);
  outp.y = pk2(av.z + bv.z, av.w + bv.w);
  *(uint2*)(y + i) = outp;
}

// ---------------------------------------------------------------------------
extern "C" void kernel_launch(void* const* d_in, const int* in_sizes, int n_in,
                              void* d_out, int out_size, void* d_ws, size_t ws_size,
                              hipStream_t stream) {
  const float* xyz     = (const float*)d_in[0];
  const float* rgb     = (const float*)d_in[1];
  const float* centers = (const float*)d_in[2];
  const float* cfeats  = (const float*)d_in[3];
  const float* rel_W1  = (const float*)d_in[4];
  const float* rel_b1  = (const float*)d_in[5];
  const float* rel_g   = (const float*)d_in[6];
  const float* rel_bb  = (const float*)d_in[7];
  const float* rel_W2  = (const float*)d_in[8];
  const float* rel_b2  = (const float*)d_in[9];
  const float* mlp_W1  = (const float*)d_in[10];
  const float* mlp_b1  = (const float*)d_in[11];
  const float* mlp_g   = (const float*)d_in[12];
  const float* mlp_bb  = (const float*)d_in[13];
  const float* mlp_W2  = (const float*)d_in[14];
  const float* mlp_b2  = (const float*)d_in[15];
  const float* fc_W1   = (const float*)d_in[16];
  const float* fc_b1   = (const float*)d_in[17];
  const float* fc_g    = (const float*)d_in[18];
  const float* fc_bb   = (const float*)d_in[19];
  const float* fc_W2   = (const float*)d_in[20];
  const float* fc_b2   = (const float*)d_in[21];

  size_t off = 0;
  auto carve = [&](size_t bytes) -> char* {
    char* p = (char*)d_ws + off;
    off += (bytes + 255) & ~(size_t)255;
    return p;
  };
  int*    idx   = (int*)   carve((size_t)BN * 3 * 4);
  float*  wgt   = (float*) carve((size_t)BN * 3 * 4);
  __bf16* wRel2 = (__bf16*)carve((size_t)65536 * 2);
  __bf16* wMlp1 = (__bf16*)carve((size_t)65536 * 2);
  __bf16* wMlp2 = (__bf16*)carve((size_t)65536 * 2);
  __bf16* wFc1  = (__bf16*)carve((size_t)65536 * 2);
  __bf16* wFc2  = (__bf16*)carve((size_t)65536 * 2);
  float*  bufX  = (float*) carve((size_t)BN * 256 * 4);  // skip (f32)
  float*  bufH  = (float*) carve((size_t)BN * 256 * 4);  // hidden (f32)
  __bf16* actA  = (__bf16*)carve((size_t)BN * 256 * 2);
  __bf16* actB  = (__bf16*)carve((size_t)BN * 256 * 2);

  // 0) weights -> bf16, transposed to [n][k] fragment-feed layout
  cvt_bf16_t_kernel<<<256, 256, 0, stream>>>(rel_W2, wRel2);
  cvt_bf16_t_kernel<<<256, 256, 0, stream>>>(mlp_W1, wMlp1); // first 256 rows
  cvt_bf16_t_kernel<<<256, 256, 0, stream>>>(mlp_W2, wMlp2);
  cvt_bf16_t_kernel<<<256, 256, 0, stream>>>(fc_W1,  wFc1);
  cvt_bf16_t_kernel<<<256, 256, 0, stream>>>(fc_W2,  wFc2);

  // 1) KNN(3) + IDW weights
  knn3_kernel<<<dim3(Nc / 256, Bc), 256, 0, stream>>>(xyz, centers, idx, wgt);

  // 2) rel layer-1 + LN + GELU, weighted-combined over K -> actA (bf16)
  rel_hidden_kernel<<<2048, 256, 0, stream>>>(xyz, centers, idx, wgt,
                                              rel_W1, rel_b1, rel_g, rel_bb, actA);

  // 3) rela_feats = G @ rel_W2 + rel_b2 -> bufX
  gemm_bf16_256<<<BN / 32, 256, 0, stream>>>(actA, wRel2, rel_b2, bufX, BN);

  // 4) skip = rela_feats + IDW-gathered center_feats; bf16 copy -> actB
  combine_kernel<<<BN, 256, 0, stream>>>(cfeats, idx, wgt, bufX, actB);

  // 5) mlp layer-1: h = skip @ W1[:256] + b1, then += rgb @ W1[256:259]
  gemm_bf16_256<<<BN / 32, 256, 0, stream>>>(actB, wMlp1, mlp_b1, bufH, BN);
  rgb_tail_kernel<<<BN, 256, 0, stream>>>(rgb, mlp_W1, bufH);
  ln_gelu_kernel<<<2048, 256, 0, stream>>>(bufH, mlp_g, mlp_bb, actA);

  // 6) mlp layer-2: h2 = act @ W2 + b2; y = skip + h2 -> actA (bf16)
  gemm_bf16_256<<<BN / 32, 256, 0, stream>>>(actA, wMlp2, mlp_b2, bufH, BN);
  residual_kernel<<<BN * 256 / 1024, 256, 0, stream>>>(bufX, bufH, actA);

  // 7) fc: (y @ W1 + b1) -> LN/GELU -> (@ W2 + b2) -> d_out (f32)
  gemm_bf16_256<<<BN / 32, 256, 0, stream>>>(actA, wFc1, fc_b1, bufH, BN);
  ln_gelu_kernel<<<2048, 256, 0, stream>>>(bufH, fc_g, fc_bb, actB);
  gemm_bf16_256<<<BN / 32, 256, 0, stream>>>(actB, wFc2, fc_b2, (float*)d_out, BN);
}
